// MoELayer_26405458936367
// MI455X (gfx1250) — matrix operand order repaired and checked by reading, not compiled, not run on hardware
//
#include <hip/hip_runtime.h>
#include <hip/hip_bf16.h>

// ---------------------------------------------------------------------------
// MoE (top-2 of 24 experts), d_model=1024, d_ff=4096, T=4096 tokens.
// Pipeline: init -> (weights f32 -> bf16, swizzled to WMMA B-fragment layout)
//           -> gate (top-2 softmax, expert lists) -> fused expert FFN
//              (bf16 WMMA GEMM1 -> exact GELU -> bf16 WMMA GEMM2 -> scatter).
// BM=32 token rows per block: each 1KB swizzled B fragment (2x b128 per lane)
// feeds TWO v_wmma_f32_16x16x32_bf16, i.e. 1 wmma : 1 b128 steady state.
// ---------------------------------------------------------------------------

#define T_TOK 4096
#define DM    1024
#define DF    4096
#define NE    24
#define BM    32      // token rows per FFN block (2 M-tiles)
#define FC    128     // ff-chunk width streamed through LDS
#define XLD   1032    // padded LDS stride (bf16 elems) for X tile
#define HLD   136     // padded LDS stride (bf16 elems) for H chunk

typedef __attribute__((ext_vector_type(16))) __bf16          v16bf;
typedef __attribute__((ext_vector_type(16))) unsigned short  v16u;
typedef __attribute__((ext_vector_type(8)))  float           v8f;

__device__ __forceinline__ unsigned short f2bf(float f) {
  unsigned int u = __builtin_bit_cast(unsigned int, f);
  unsigned int r = u + 0x7FFFu + ((u >> 16) & 1u);   // round-to-nearest-even
  return (unsigned short)(r >> 16);
}
__device__ __forceinline__ unsigned int pack_bf16(float a, float b) {
  return (unsigned int)f2bf(a) | ((unsigned int)f2bf(b) << 16);
}

// A-fragment (16x32 bf16) from LDS, contiguous-K layout matching B:
// lane holds row M=lane&15; elements 0..15 = K=ko..ko+15;
// ko = (lane>=16)?16:0.  One contiguous 32B run -> ds_load_b128 pairs.
__device__ __forceinline__ v16bf load_a(const unsigned short* base, int ld, int lane) {
  const int m  = lane & 15;
  const int ko = (lane >> 4) << 4;
  const unsigned short* p = base + m * ld + ko;
  v16u t;
#pragma unroll
  for (int j = 0; j < 16; ++j) t[j] = p[j];
  return __builtin_bit_cast(v16bf, t);
}

// B-fragment from pre-swizzled bf16 weights: lane reads 32 contiguous bytes.
__device__ __forceinline__ v16bf load_b_swz(const unsigned short* frag, int lane) {
  const v16u t = *(const v16u*)(frag + lane * 16);   // 32B aligned -> 2x b128
  return __builtin_bit_cast(v16bf, t);
}

// Fallback B-fragment from row-major f32 weights (strided, convert-on-load).
__device__ __forceinline__ v16bf load_b_cvt(const float* w, int ld, int lane) {
  const int n  = lane & 15;
  const int kb = (lane >> 4) << 4;
  const float* p = w + (size_t)kb * ld + n;
  v16u t;
#pragma unroll
  for (int j = 0; j < 16; ++j) t[j] = f2bf(p[(size_t)j * ld]);
  return __builtin_bit_cast(v16bf, t);
}

__device__ __forceinline__ float gelu_exact(float v) {
  return 0.5f * v * (1.0f + erff(v * 0.70710678118654752f));
}

// ---------------------------------------------------------------------------
// Kernel 0: zero output accumulator and per-expert counters.
// ---------------------------------------------------------------------------
__global__ void moe_init(float* __restrict__ out, int* __restrict__ counts, int n) {
  int i = blockIdx.x * blockDim.x + threadIdx.x;
  if (i < n)  out[i] = 0.0f;
  if (i < NE) counts[i] = 0;
}

// ---------------------------------------------------------------------------
// Kernel 1: weight convert + swizzle.  src: [E][K][N] f32 row-major.
// dst: per expert, fragments of 512 bf16 (one 32x16 K-major B tile) laid out
// element (lane, j) = (k = kk*32 + ((lane>=16)?16:0) + j, n = nt*16 + lane&15),
// fragment address = ((nt * (K/32)) + kk) * 512.
// Block = one 32(k) x 128(n) tile; coalesced f32-pair reads, LDS transpose,
// coalesced 32B fragment writes.
// ---------------------------------------------------------------------------
__global__ __launch_bounds__(256) void moe_cvt_swizzle(
    const float* __restrict__ src, unsigned short* __restrict__ dst,
    int K, int N) {
  const int e  = blockIdx.z;
  const int k0 = blockIdx.y * 32;
  const int n0 = blockIdx.x * 128;
  const int tid = threadIdx.x;

  __shared__ unsigned short TS[32 * HLD];   // 32 x 128 bf16 tile (padded)

  const float* srcE = src + (size_t)e * K * N;
  unsigned short* dstE = dst + (size_t)e * K * N;

  // read 32x128 f32 tile (pairs along n, coalesced), pack to bf16 in LDS
#pragma unroll
  for (int p = 0; p < 8; ++p) {
    const int pid = tid + p * 256;          // 2048 pairs
    const int k  = pid >> 6;                // 64 pairs per k-row
    const int n  = (pid & 63) << 1;
    const float* sp = srcE + (size_t)(k0 + k) * N + n0 + n;
    *(unsigned int*)&TS[k * HLD + n] = pack_bf16(sp[0], sp[1]);
  }
  __syncthreads();

  // emit 8 fragments; thread (fragl = tid>>5, lane = tid&31) writes 32B
  const int fragl = tid >> 5;
  const int lane  = tid & 31;
  const int nn    = fragl * 16 + (lane & 15);
  const int kb    = (lane >> 4) << 4;
  v16u t;
#pragma unroll
  for (int j = 0; j < 16; ++j) t[j] = TS[(kb + j) * HLD + nn];

  unsigned short* fp = dstE +
      (((size_t)(n0 / 16 + fragl) * (K / 32) + (k0 / 32)) << 9);
  *(v16u*)(fp + lane * 16) = t;             // coalesced 1KB per fragment
}

// ---------------------------------------------------------------------------
// Kernel 2: gating. One wave32 per token: scores = x@gate_w + gate_b,
// top-2, softmax over the two logits, atomic slot append to expert lists.
// ---------------------------------------------------------------------------
__global__ __launch_bounds__(32) void moe_gate(
    const float* __restrict__ x, const float* __restrict__ gw,
    const float* __restrict__ gb, int* __restrict__ counts,
    int* __restrict__ etok, float* __restrict__ egate) {
  const int t    = blockIdx.x;
  const int lane = threadIdx.x;

  float acc[NE];
#pragma unroll
  for (int e = 0; e < NE; ++e) acc[e] = 0.0f;

  const float* xt = x + (size_t)t * DM;
  for (int d = lane; d < DM; d += 32) {
    const float xv = xt[d];
#pragma unroll
    for (int e = 0; e < NE; ++e) acc[e] += xv * gw[d * NE + e];
  }
#pragma unroll
  for (int e = 0; e < NE; ++e)
    for (int off = 16; off > 0; off >>= 1)
      acc[e] += __shfl_xor(acc[e], off, 32);

  if (lane == 0) {
    int i0 = 0; float v0 = -1e30f;
#pragma unroll
    for (int e = 0; e < NE; ++e) {
      const float s = acc[e] + gb[e];
      acc[e] = s;
      if (s > v0) { v0 = s; i0 = e; }      // first-occurrence tie-break
    }
    int i1 = 0; float v1 = -1e30f;
#pragma unroll
    for (int e = 0; e < NE; ++e)
      if (e != i0 && acc[e] > v1) { v1 = acc[e]; i1 = e; }

    const float e1 = expf(v1 - v0);        // v0 is the max
    const float inv = 1.0f / (1.0f + e1);
    const float p0 = inv, p1 = e1 * inv;

    int s0 = atomicAdd(&counts[i0], 1);
    etok [i0 * T_TOK + s0] = t;
    egate[i0 * T_TOK + s0] = p0;
    int s1 = atomicAdd(&counts[i1], 1);
    etok [i1 * T_TOK + s1] = t;
    egate[i1 * T_TOK + s1] = p1;
  }
}

// ---------------------------------------------------------------------------
// Kernel 3: fused expert FFN. Block = (expert e, 32-token tile), 8 waves.
// Streams d_ff in 128-wide chunks: GEMM1 (K=1024) -> +b1 -> GELU -> bf16 LDS
// -> GEMM2 (K=128/chunk) in VGPR accumulators -> *gate, +b2, atomic scatter.
// Two M-tiles per wave: each B fragment is consumed by 2 WMMAs.
// SWZ=true : B from pre-swizzled bf16 fragments (2x b128 / fragment / lane).
// SWZ=false: B converted on the fly from f32 (workspace-too-small fallback).
// ---------------------------------------------------------------------------
template <bool SWZ>
__global__ __launch_bounds__(256) void moe_ffn(
    const float* __restrict__ x,
    const float* __restrict__ w1f, const float* __restrict__ b1,
    const float* __restrict__ w2f, const float* __restrict__ b2,
    const unsigned short* __restrict__ w1s,
    const unsigned short* __restrict__ w2s,
    const int* __restrict__ counts, const int* __restrict__ etok,
    const float* __restrict__ egate, float* __restrict__ out) {
  const int e    = blockIdx.y;
  const int cnt  = counts[e];
  const int row0 = blockIdx.x * BM;
  if (row0 >= cnt) return;                 // uniform early-exit

  __shared__ unsigned short Xs[BM * XLD];  // 66.0 KB  x-tile (bf16)
  __shared__ unsigned short Hs[BM * HLD];  //  8.7 KB  gelu chunk (bf16)
  __shared__ int   toks[BM];
  __shared__ float gates[BM];

  const int tid  = threadIdx.x;
  const int lane = tid & 31;
  const int wv   = tid >> 5;               // 8 waves

  if (tid < BM) {
    const int r = row0 + tid;
    toks[tid]  = (r < cnt) ? etok [e * T_TOK + r] : -1;
    gates[tid] = (r < cnt) ? egate[e * T_TOK + r] : 0.0f;
  }
  __syncthreads();

  // gather X tile, f32 -> bf16 packed pairs (padding rows are zero)
  for (int idx = tid; idx < BM * DM / 2; idx += 256) {
    const int r  = idx >> 9;               // / (1024/2)
    const int d  = (idx & (DM / 2 - 1)) << 1;
    const int tk = toks[r];
    unsigned int pk = 0u;
    if (tk >= 0) {
      const float* xp = x + (size_t)tk * DM + d;
      pk = pack_bf16(xp[0], xp[1]);
    }
    *(unsigned int*)&Xs[r * XLD + d] = pk;
  }

  const float*          w1fe = w1f + (size_t)e * DM * DF;
  const float*          w2fe = w2f + (size_t)e * DF * DM;
  const unsigned short* w1se = w1s + (size_t)e * DM * DF;
  const unsigned short* w2se = w2s + (size_t)e * DF * DM;

  v8f accY[2][8] = {};                     // wave wv owns out cols [128*wv, +128)

  for (int c = 0; c < DF / FC; ++c) {
    const int fc = c * FC;

    // ---- GEMM1: wave wv computes H cols [fc+16*wv, +16) x 2 M-tiles ----
    v8f h0 = {}, h1 = {};
    const int n1 = fc + wv * 16;
    for (int kk = 0; kk < DM / 32; ++kk) {
      v16bf a0 = load_a(Xs + kk * 32, XLD, lane);
      v16bf a1 = load_a(Xs + 16 * XLD + kk * 32, XLD, lane);
      v16bf b;
      if (SWZ)
        b = load_b_swz(w1se + (((size_t)(c * 8 + wv) * (DM / 32) + kk) << 9), lane);
      else
        b = load_b_cvt(w1fe + (size_t)(kk * 32) * DF + n1, DF, lane);
      h0 = __builtin_amdgcn_wmma_f32_16x16x32_bf16(false, a0, false, b,
                                                   (short)0, h0, false, false);
      h1 = __builtin_amdgcn_wmma_f32_16x16x32_bf16(false, a1, false, b,
                                                   (short)0, h1, false, false);
    }
    const float bias1 = b1[e * DF + n1 + (lane & 15)];
    __syncthreads();                       // previous chunk's GEMM2 done with Hs
#pragma unroll
    for (int r = 0; r < 8; ++r) {
      const int m = r + ((lane >> 4) << 3);      // C layout: VGPR r -> row
      const int col = wv * 16 + (lane & 15);
      Hs[m * HLD + col]              = f2bf(gelu_exact(h0[r] + bias1));
      Hs[(16 + m) * HLD + col]       = f2bf(gelu_exact(h1[r] + bias1));
    }
    __syncthreads();                       // Hs chunk ready

    // ---- GEMM2: Y[:, 128*wv ..) += H_chunk @ w2e[fc..fc+128, :] ----
#pragma unroll
    for (int kk = 0; kk < 4; ++kk) {
      v16bf af0 = load_a(Hs + kk * 32, HLD, lane);
      v16bf af1 = load_a(Hs + 16 * HLD + kk * 32, HLD, lane);
#pragma unroll
      for (int t = 0; t < 8; ++t) {
        v16bf b;
        if (SWZ)
          b = load_b_swz(
              w2se + (((size_t)(wv * 8 + t) * (DF / 32) + (c * 4 + kk)) << 9), lane);
        else
          b = load_b_cvt(w2fe + (size_t)(fc + kk * 32) * DM + wv * 128 + t * 16,
                         DM, lane);
        accY[0][t] = __builtin_amdgcn_wmma_f32_16x16x32_bf16(
            false, af0, false, b, (short)0, accY[0][t], false, false);
        accY[1][t] = __builtin_amdgcn_wmma_f32_16x16x32_bf16(
            false, af1, false, b, (short)0, accY[1][t], false, false);
      }
    }
  }

  // ---- epilogue: out[tok, :] += gate * (Y + b2) (2 atomic adds / element) ----
#pragma unroll
  for (int t = 0; t < 8; ++t) {
    const int n2 = wv * 128 + t * 16 + (lane & 15);
    const float bias2 = b2[e * DM + n2];
#pragma unroll
    for (int mt = 0; mt < 2; ++mt) {
#pragma unroll
      for (int r = 0; r < 8; ++r) {
        const int m  = mt * 16 + r + ((lane >> 4) << 3);
        const int tk = toks[m];
        if (tk >= 0)
          atomicAdd(&out[(size_t)tk * DM + n2],
                    gates[m] * (accY[mt][t][r] + bias2));
      }
    }
  }
}

// ---------------------------------------------------------------------------
// Host launcher. Inputs: x, gate_w, gate_b, w1, b1, w2, b2 (all f32).
// Workspace: [counts 128B | etok 384KB | egate 384KB | @1MB: w1 bf16 swz
//             192MB | w2 bf16 swz 192MB].  Falls back to convert-on-load FFN
// if ws_size cannot hold the swizzled weights.
// ---------------------------------------------------------------------------
extern "C" void kernel_launch(void* const* d_in, const int* in_sizes, int n_in,
                              void* d_out, int out_size, void* d_ws, size_t ws_size,
                              hipStream_t stream) {
  (void)in_sizes; (void)n_in; (void)out_size;
  const float* x      = (const float*)d_in[0];
  const float* gate_w = (const float*)d_in[1];
  const float* gate_b = (const float*)d_in[2];
  const float* w1     = (const float*)d_in[3];
  const float* b1     = (const float*)d_in[4];
  const float* w2     = (const float*)d_in[5];
  const float* b2     = (const float*)d_in[6];
  float*       out    = (float*)d_out;

  char*  ws     = (char*)d_ws;
  int*   counts = (int*)ws;
  int*   etok   = (int*)(ws + 128);
  float* egate  = (float*)(ws + 128 + (size_t)NE * T_TOK * sizeof(int));

  const size_t wbytes = (size_t)NE * DM * DF * sizeof(unsigned short); // 192MB
  unsigned short* w1s = (unsigned short*)(ws + (1u << 20));
  unsigned short* w2s = (unsigned short*)(ws + (1u << 20) + wbytes);
  const bool fast = ws_size >= (1u << 20) + 2 * wbytes;

  const int n = T_TOK * DM;
  moe_init<<<(n + 255) / 256, 256, 0, stream>>>(out, counts, n);
  moe_gate<<<T_TOK, 32, 0, stream>>>(x, gate_w, gate_b, counts, etok, egate);

  dim3 grid(T_TOK / BM, NE);               // 128 tiles x 24 experts, early-exit
  if (fast) {
    moe_cvt_swizzle<<<dim3(DF / 128, DM / 32, NE), 256, 0, stream>>>(w1, w1s, DM, DF);
    moe_cvt_swizzle<<<dim3(DM / 128, DF / 32, NE), 256, 0, stream>>>(w2, w2s, DF, DM);
    moe_ffn<true><<<grid, 256, 0, stream>>>(x, w1, b1, w2, b2, w1s, w2s,
                                            counts, etok, egate, out);
  } else {
    moe_ffn<false><<<grid, 256, 0, stream>>>(x, w1, b1, w2, b2,
                                             (const unsigned short*)nullptr,
                                             (const unsigned short*)nullptr,
                                             counts, etok, egate, out);
  }
}
